// Similarity_54975581389439
// MI455X (gfx1250) — matrix operand order repaired and checked by reading, compiled, not run on hardware
//
#include <hip/hip_runtime.h>
#include <hip/hip_bf16.h>

typedef _Float16 half_t;
typedef __attribute__((ext_vector_type(16))) _Float16 v16h;
typedef __attribute__((ext_vector_type(8)))  _Float16 v8h_t;
typedef __attribute__((ext_vector_type(8)))  float    v8f;

#define BN_EPS 1e-5f

// ---------------------------------------------------------------------------
// Fold BN into per-channel scale/shift and repack conv weights to f16 with
// K layout k = (ky*3+kx)*Cin + ci  (so every 32-wide K chunk is one filter tap
// with contiguous input channels -> contiguous LDS B-fragment loads).
// ---------------------------------------------------------------------------
__global__ void prep_layer_kernel(const float* __restrict__ W,
                                  const float* __restrict__ b,
                                  const float* __restrict__ beta,
                                  const float* __restrict__ gamma,
                                  const float* __restrict__ mean,
                                  const float* __restrict__ var,
                                  half_t* __restrict__ wt,
                                  float* __restrict__ scale,
                                  float* __restrict__ shift,
                                  int Cin, int Cout, int Kpad) {
  int total = Cout * Kpad;
  for (int idx = threadIdx.x + blockIdx.x * blockDim.x; idx < total;
       idx += blockDim.x * gridDim.x) {
    int co = idx / Kpad, k = idx % Kpad;
    float v = 0.0f;
    if (k < 9 * Cin) {
      int r = k / Cin, ci = k - r * Cin;
      int ky = r / 3, kx = r - ky * 3;
      v = W[((co * Cin + ci) * 3 + ky) * 3 + kx];
    }
    wt[idx] = (half_t)v;
  }
  for (int co = threadIdx.x + blockIdx.x * blockDim.x; co < Cout;
       co += blockDim.x * gridDim.x) {
    float inv = gamma[co] * rsqrtf(var[co] + BN_EPS);
    scale[co] = inv;
    shift[co] = beta[co] + (b[co] - mean[co]) * inv;
  }
}

// ---------------------------------------------------------------------------
// Layer 0: Cin=1, K=9 -> direct conv (too skinny for WMMA). f32 in, f16 out,
// fused scale/shift + ReLU. 16x16 spatial tile with 18x18 LDS halo.
// ---------------------------------------------------------------------------
__global__ __launch_bounds__(256) void conv_first_kernel(
    const float* __restrict__ in, half_t* __restrict__ out,
    const half_t* __restrict__ wt, const float* __restrict__ scale,
    const float* __restrict__ shift, int H, int W) {
  __shared__ float tile[18 * 18];
  int x0 = blockIdx.x * 16, y0 = blockIdx.y * 16, n = blockIdx.z;
  const float* inp = in + (size_t)n * H * W;
  int tid = threadIdx.y * 16 + threadIdx.x;
  for (int i = tid; i < 18 * 18; i += 256) {
    int sy = i / 18, sx = i - sy * 18;
    int gy = y0 + sy - 1, gx = x0 + sx - 1;
    tile[i] = (gy >= 0 && gy < H && gx >= 0 && gx < W) ? inp[(size_t)gy * W + gx] : 0.0f;
  }
  __syncthreads();
  int x = x0 + threadIdx.x, y = y0 + threadIdx.y;
  if (x >= W || y >= H) return;
  float v[9];
#pragma unroll
  for (int r = 0; r < 9; r++) {
    int ky = r / 3, kx = r - ky * 3;
    v[r] = tile[(threadIdx.y + ky) * 18 + threadIdx.x + kx];
  }
  for (int co = 0; co < 32; co++) {
    float acc = 0.0f;
#pragma unroll
    for (int r = 0; r < 9; r++) acc += v[r] * (float)wt[co * 32 + r];
    acc = fmaxf(acc * scale[co] + shift[co], 0.0f);
    out[(((size_t)n * 32 + co) * H + y) * W + x] = (half_t)acc;
  }
}

// ---------------------------------------------------------------------------
// Implicit-GEMM conv via V_WMMA_F32_16X16X32_F16, compile-time (CIN, COUT).
//   Per block: 32 output pixels (4 rows x 8 cols) for all COUT channels.
//   Waves = COUT/16; each wave owns one 16-row M tile and BOTH 16-pixel N
//   tiles (two f32 accumulators) so each A fragment feeds two WMMAs.
//   K = CIN*9 fully unrolled in chunks of 32 (KC = 9 or 18 at compile time);
//   chunk -> (ky,kx,ci0) all fold to immediates.
// EXEC is all-ones through every WMMA; edge masking only at the stores.
// ---------------------------------------------------------------------------
template <int CIN, int COUT>
__global__ __launch_bounds__(128) void conv_wmma_kernel(
    const half_t* __restrict__ in, half_t* __restrict__ out,
    const half_t* __restrict__ wt, const float* __restrict__ scale,
    const float* __restrict__ shift, int H, int W) {
  constexpr int K = CIN * 9;
  constexpr int KC = K / 32;  // 9 or 18
  __shared__ __attribute__((aligned(32))) half_t ldsin[6 * 10 * CIN];

  int x0 = blockIdx.x * 8, y0 = blockIdx.y * 4, n = blockIdx.z;
  int tid = threadIdx.x;
  const int nthreads = (COUT / 16) * 32;
  const half_t* inp = in + (size_t)n * CIN * H * W;

  // cooperative halo load: rows of 10 contiguous x per (ci, sy) -> coalesced
  constexpr int total = CIN * 6 * 10;
  for (int i = tid; i < total; i += nthreads) {
    int sx = i % 10;
    int t = i / 10;
    int sy = t % 6;
    int ci = t / 6;
    int gy = y0 + sy - 1, gx = x0 + sx - 1;
    half_t v = (half_t)0.0f;
    if (gy >= 0 && gy < H && gx >= 0 && gx < W)
      v = inp[((size_t)ci * H + gy) * W + gx];
    ldsin[(sy * 10 + sx) * CIN + ci] = v;
  }
  __syncthreads();

  int wave = tid >> 5, lane = tid & 31;
  int hiA = (lane & 16) ? 16 : 0;   // A: K half-set select
  int lo16 = (lane & 16) ? 16 : 0;  // B: K half-set select
  int m = lane & 15;                // A row / C column index
  int dy = m >> 3, dx = m & 7;      // pixel owned in N tile 0 (tile 1: +2 rows)
  const half_t* wrow = wt + (size_t)(wave * 16 + m) * K;
  const half_t* bbase = &ldsin[(dy * 10 + dx) * CIN + lo16];

  v8f c0 = {}, c1 = {};
#pragma unroll
  for (int kc = 0; kc < KC; kc++) {
    const int kb = kc * 32;
    const int r = kb / CIN;  // exact; folds to an immediate
    const int ci0 = kb - r * CIN;
    const int ky = r / 3, kx = r - ky * 3;

    // A fragment: two contiguous 16B runs of this lane's weight row
    v8h_t a0 = *(const v8h_t*)(wrow + kb + hiA);
    v8h_t a1 = *(const v8h_t*)(wrow + kb + 8 + hiA);
    v16h A;
#pragma unroll
    for (int e = 0; e < 8; e++) { A[e] = a0[e]; A[8 + e] = a1[e]; }

    // B fragments: 16 contiguous input channels at this pixel's tap;
    // N tile 1 is the same pixel shifted down 2 rows (+20*CIN halves).
    const int boff = (ky * 10 + kx) * CIN + ci0;
    const v16h B0 = *(const v16h*)(bbase + boff);
    const v16h B1 = *(const v16h*)(bbase + boff + 20 * CIN);

    c0 = __builtin_amdgcn_wmma_f32_16x16x32_f16(false, A, false, B0, (short)0, c0,
                                                false, false);
    c1 = __builtin_amdgcn_wmma_f32_16x16x32_f16(false, A, false, B1, (short)0, c1,
                                                false, false);
  }

  int x = x0 + dx;
  int ya = y0 + dy;      // rows for accumulator c0 (pixels 0..15)
  int yb = ya + 2;       // rows for accumulator c1 (pixels 16..31)
  int mbase = wave * 16 + ((lane & 16) ? 8 : 0);
  if (x < W) {
#pragma unroll
    for (int rr = 0; rr < 8; rr++) {
      int co = mbase + rr;
      float sc = scale[co], sh = shift[co];
      size_t base = ((size_t)n * COUT + co) * H;
      if (ya < H) {
        float v = fmaxf(c0[rr] * sc + sh, 0.0f);
        out[(base + ya) * W + x] = (half_t)v;
      }
      if (yb < H) {
        float v = fmaxf(c1[rr] * sc + sh, 0.0f);
        out[(base + yb) * W + x] = (half_t)v;
      }
    }
  }
}

// ---------------------------------------------------------------------------
__global__ void maxpool_kernel(const half_t* __restrict__ in,
                               half_t* __restrict__ out, long long total,
                               int Ho, int Wo, int H, int W) {
  long long i = (long long)blockIdx.x * blockDim.x + threadIdx.x;
  if (i >= total) return;
  int xo = (int)(i % Wo);
  long long t = i / Wo;
  int yo = (int)(t % Ho);
  long long nc = t / Ho;
  const half_t* p = in + (nc * H + (long long)yo * 2) * W + (long long)xo * 2;
  float a = (float)p[0], b = (float)p[1], cc = (float)p[W], d = (float)p[W + 1];
  out[i] = (half_t)fmaxf(fmaxf(a, b), fmaxf(cc, d));
}

// ---------------------------------------------------------------------------
__global__ __launch_bounds__(256) void asum_kernel(const half_t* __restrict__ fa,
                                                   float* __restrict__ asum) {
  __shared__ float red[256];
  int b = blockIdx.x;
  const half_t* p = fa + (size_t)b * 64 * 32 * 32;
  float s = 0.0f;
  for (int i = threadIdx.x; i < 64 * 32 * 32; i += 256) {
    float v = (float)p[i];
    s += v * v;
  }
  red[threadIdx.x] = s;
  __syncthreads();
  for (int st = 128; st > 0; st >>= 1) {
    if (threadIdx.x < st) red[threadIdx.x] += red[threadIdx.x + st];
    __syncthreads();
  }
  if (threadIdx.x == 0) asum[b] = red[0];
}

__global__ void sq_kernel(const half_t* __restrict__ fb, float* __restrict__ sq) {
  int i = blockIdx.x * blockDim.x + threadIdx.x;
  if (i >= 4 * 50 * 50) return;
  int b = i / 2500;
  int yx = i - b * 2500;
  const half_t* p = fb + (size_t)b * 64 * 2500 + yx;
  float s = 0.0f;
  for (int c = 0; c < 64; c++) {
    float v = (float)p[(size_t)c * 2500];
    s += v * v;
  }
  sq[i] = s;
}

__global__ __launch_bounds__(256) void ncc_kernel(
    const half_t* __restrict__ fa, const half_t* __restrict__ fb,
    const float* __restrict__ sq, const float* __restrict__ asum,
    float* __restrict__ out) {
  __shared__ float redc[256];
  __shared__ float redb[256];
  int blk = blockIdx.x;
  int b = blk / 361;
  int s = blk - b * 361;
  int sy = s / 19, sx = s - sy * 19;
  const half_t* pa = fa + (size_t)b * 64 * 1024;
  const half_t* pb = fb + (size_t)b * 64 * 2500;
  float cr = 0.0f;
  for (int i = threadIdx.x; i < 65536; i += 256) {
    int c = i >> 10;
    int rem = i & 1023;
    int y = rem >> 5, x = rem & 31;
    cr += (float)pa[(c << 10) + rem] *
          (float)pb[(size_t)c * 2500 + (y + sy) * 50 + (x + sx)];
  }
  float bs = 0.0f;
  const float* ps = sq + b * 2500;
  for (int j = threadIdx.x; j < 1024; j += 256) {
    int y = j >> 5, x = j & 31;
    bs += ps[(y + sy) * 50 + x + sx];
  }
  redc[threadIdx.x] = cr;
  redb[threadIdx.x] = bs;
  __syncthreads();
  for (int st = 128; st > 0; st >>= 1) {
    if (threadIdx.x < st) {
      redc[threadIdx.x] += redc[threadIdx.x + st];
      redb[threadIdx.x] += redb[threadIdx.x + st];
    }
    __syncthreads();
  }
  if (threadIdx.x == 0) out[blk] = redc[0] * rsqrtf(asum[b] * redb[0]);
}

// ---------------------------------------------------------------------------
extern "C" void kernel_launch(void* const* d_in, const int* in_sizes, int n_in,
                              void* d_out, int out_size, void* d_ws, size_t ws_size,
                              hipStream_t stream) {
  (void)in_sizes; (void)n_in; (void)out_size; (void)ws_size;
  const float* inA = (const float*)d_in[0]; // [4,1,512,512]
  const float* inB = (const float*)d_in[1]; // [4,1,800,800]

  // Workspace layout (halves unless noted). Largest activation: 4*32*800*800.
  const size_t PING = (size_t)4 * 32 * 800 * 800; // 81,920,000 halves
  half_t* ping = (half_t*)d_ws;
  half_t* pong = ping + PING;
  half_t* fa = pong + PING;                       // 4*64*32*32
  half_t* fb = fa + (size_t)4 * 64 * 32 * 32;     // 4*64*50*50
  half_t* wt = fb + (size_t)4 * 64 * 50 * 50;     // 212,992 halves total
  float* scale = (float*)(wt + 212992);
  float* shift = scale + 448;
  float* asum = shift + 448;
  float* sqbuf = asum + 4;

  static const int CIN[8] = {1, 32, 32, 64, 64, 64, 64, 64};
  static const int COUT[8] = {32, 32, 64, 64, 64, 64, 64, 64};
  static const size_t WOFF[8] = {0, 1024, 10240, 28672, 65536, 102400, 139264, 176128};
  static const int CHOFF[8] = {0, 32, 64, 128, 192, 256, 320, 384};

  // Weight / BN prep (params flattened: per-layer W,b,beta,gamma,mean,var)
  for (int li = 0; li < 8; li++) {
    const float* Wp = (const float*)d_in[2 + 6 * li + 0];
    const float* bp = (const float*)d_in[2 + 6 * li + 1];
    const float* betap = (const float*)d_in[2 + 6 * li + 2];
    const float* gammap = (const float*)d_in[2 + 6 * li + 3];
    const float* meanp = (const float*)d_in[2 + 6 * li + 4];
    const float* varp = (const float*)d_in[2 + 6 * li + 5];
    int Kpad = (li == 0) ? 32 : CIN[li] * 9;
    prep_layer_kernel<<<64, 256, 0, stream>>>(Wp, bp, betap, gammap, meanp, varp,
                                              wt + WOFF[li], scale + CHOFF[li],
                                              shift + CHOFF[li], CIN[li], COUT[li],
                                              Kpad);
  }

  // Feature towers
  for (int dom = 0; dom < 2; dom++) {
    const float* src = dom ? inB : inA;
    int H = dom ? 800 : 512, W = H;
    half_t* feat = dom ? fb : fa;

    dim3 g0((W + 15) / 16, (H + 15) / 16, 4);
    conv_first_kernel<<<g0, dim3(16, 16), 0, stream>>>(src, ping, wt + WOFF[0],
                                                       scale, shift, H, W);
    half_t* bufs[2] = {ping, pong};
    int cb = 0;
    for (int li = 1; li < 8; li++) {
      dim3 g((W + 7) / 8, (H + 3) / 4, 4);
      const half_t* s = bufs[cb];
      half_t* d = bufs[1 - cb];
      const half_t* w = wt + WOFF[li];
      const float* sc = scale + CHOFF[li];
      const float* sh = shift + CHOFF[li];
      if (CIN[li] == 32 && COUT[li] == 32) {
        conv_wmma_kernel<32, 32><<<g, 64, 0, stream>>>(s, d, w, sc, sh, H, W);
      } else if (CIN[li] == 32 && COUT[li] == 64) {
        conv_wmma_kernel<32, 64><<<g, 128, 0, stream>>>(s, d, w, sc, sh, H, W);
      } else {
        conv_wmma_kernel<64, 64><<<g, 128, 0, stream>>>(s, d, w, sc, sh, H, W);
      }
      cb ^= 1;
      if (li == 1 || li == 3 || li == 5 || li == 7) {
        int Ho = H / 2, Wo = W / 2;
        half_t* dst = (li == 7) ? feat : bufs[1 - cb];
        long long total = (long long)4 * COUT[li] * Ho * Wo;
        maxpool_kernel<<<(unsigned)((total + 255) / 256), 256, 0, stream>>>(
            bufs[cb], dst, total, Ho, Wo, H, W);
        if (li != 7) cb ^= 1;
        H = Ho;
        W = Wo;
      }
    }
  }

  // NCC search
  asum_kernel<<<4, 256, 0, stream>>>(fa, asum);
  sq_kernel<<<(4 * 50 * 50 + 255) / 256, 256, 0, stream>>>(fb, sqbuf);
  ncc_kernel<<<4 * 361, 256, 0, stream>>>(fa, fb, sqbuf, asum, (float*)d_out);
}
